// GraphNet_68590627717222
// MI455X (gfx1250) — matrix operand order repaired and checked by reading, not compiled, not run on hardware
//
#include <hip/hip_runtime.h>
#include <hip/hip_bf16.h>

// ---------------------------------------------------------------- constants
#define D_      128
#define H_      128
#define NM_     100000
#define NH_     10000
#define NTOT_   (NM_ + NH_)
#define EM_     400000
#define EU_     100000
#define EI_     50000
#define ED_     100000

typedef __attribute__((ext_vector_type(16))) _Float16 v16h;
typedef __attribute__((ext_vector_type(8)))  _Float16 v8h;
typedef __attribute__((ext_vector_type(8)))  float    v8f;

struct AggPtrs { const float* S; const float* MX; const float* MN; const float* C; };

// ---------------------------------------------------------------- helpers
__device__ __forceinline__ void atomicMaxF(float* a, float v) {
    if (v >= 0.0f) atomicMax((int*)a, __float_as_int(v));
    else           atomicMin((unsigned int*)a, __float_as_uint(v));
}
__device__ __forceinline__ void atomicMinF(float* a, float v) {
    if (v >= 0.0f) atomicMin((int*)a, __float_as_int(v));
    else           atomicMax((unsigned int*)a, __float_as_uint(v));
}

__device__ __forceinline__ float node_feat(const float* mesh, const float* hyper,
                                           int node, int k) {
    return (node < NM_) ? mesh[(size_t)node * D_ + k]
                        : hyper[(size_t)(node - NM_) * D_ + k];
}

// PNA stat fetch with torch_scatter empty-segment semantics (empty -> 0)
__device__ __forceinline__ float agg_val(const AggPtrs& a, int node, int stat, int n) {
    float c = a.C[node];
    size_t i = (size_t)node * D_ + n;
    if (stat == 0) return a.S[i];
    if (stat == 1) return a.S[i] / fmaxf(c, 1.0f);
    if (stat == 2) return (c > 0.0f) ? a.MX[i] : 0.0f;
    return (c > 0.0f) ? a.MN[i] : 0.0f;
}

// ---------------------------------------------------------------- fill
__global__ __launch_bounds__(256) void fill_u32(unsigned* p, int n, unsigned v) {
    int i = blockIdx.x * blockDim.x + threadIdx.x;
    if (i < n) p[i] = v;
}

// ------------------------------------------------- weight pack (B-fragment order)
// Fragment f = ks*8 + t. Element (f, lane, e) holds W[K][N] with
//   K = ks*32 + (e<8 ? 8*(lane>>4) + e : 16 + 8*(lane>>4) + (e-8)),  N = t*16 + (lane&15)
__global__ __launch_bounds__(256) void pack_w(const float* __restrict__ W, int K,
                                              _Float16* __restrict__ Wp) {
    int idx = blockIdx.x * blockDim.x + threadIdx.x;
    int total = K * D_;
    if (idx >= total) return;
    int e  = idx & 15;
    int l  = (idx >> 4) & 31;
    int f  = idx >> 9;
    int ks = f >> 3;
    int t  = f & 7;
    int kk = ks * 32 + ((e < 8) ? 8 * (l >> 4) + e : 16 + 8 * (l >> 4) + (e - 8));
    int n  = t * 16 + (l & 15);
    Wp[idx] = (_Float16)W[(size_t)kk * D_ + n];
}

// ------------------------------------------------- input staging kernels (f16)
__global__ __launch_bounds__(256) void build_edge_x(
    const float* __restrict__ mesh, const float* __restrict__ hyper,
    const float* __restrict__ ef, const int* __restrict__ s, const int* __restrict__ r,
    int E, _Float16* __restrict__ X) {
    int idx = blockIdx.x * blockDim.x + threadIdx.x;
    int total = E * 384;
    if (idx >= total) return;
    int e = idx / 384;
    int k = idx - e * 384;
    float v;
    if (k < D_)            v = node_feat(mesh, hyper, s[e], k);
    else if (k < 2 * D_)   v = node_feat(mesh, hyper, r[e], k - D_);
    else                   v = ef[(size_t)e * D_ + (k - 2 * D_)];
    X[idx] = (_Float16)v;
}

// node_cross input: [mesh_feat(128) | a_mesh(512) | a_up(512) | a_inter(512) | a_down(512)]
__global__ __launch_bounds__(256) void build_xc(
    const float* __restrict__ mesh, AggPtrs a0, AggPtrs a1, AggPtrs a2, AggPtrs a3,
    _Float16* __restrict__ X) {
    int idx = blockIdx.x * blockDim.x + threadIdx.x;
    int total = NM_ * 2176;
    if (idx >= total) return;
    int i = idx / 2176;
    int k = idx - i * 2176;
    float v;
    if (k < D_) v = mesh[(size_t)i * D_ + k];
    else {
        int q = k - D_;
        int j = q >> 9;          // which edge-set aggregation
        int qq = q & 511;
        int stat = qq >> 7;      // sum/mean/max/min
        int n = qq & 127;
        const AggPtrs& a = (j == 0) ? a0 : (j == 1) ? a1 : (j == 2) ? a2 : a3;
        v = agg_val(a, i, stat, n);
    }
    X[idx] = (_Float16)v;
}

// generic [feat(128) | one PNA block(512)] input builder (K = 640)
__global__ __launch_bounds__(256) void build_x_single(
    const float* __restrict__ feat, AggPtrs a, int nodeOff, int rows,
    _Float16* __restrict__ X) {
    int idx = blockIdx.x * blockDim.x + threadIdx.x;
    int total = rows * 640;
    if (idx >= total) return;
    int i = idx / 640;
    int k = idx - i * 640;
    float v;
    if (k < D_) v = feat[(size_t)i * D_ + k];
    else {
        int q = k - D_;
        int stat = q >> 7;
        int n = q & 127;
        v = agg_val(a, nodeOff + i, stat, n);
    }
    X[idx] = (_Float16)v;
}

// ---------------------------------------------------------------- fused MLP+LN
// One wave per PAIR of 16-row tiles (B-fragment reuse x2):
//   Y = LN(relu(X@W1 + b1)@W2 + b2)*g + beta
// Optional: residual add to output, fused PNA atomics (sum/max/min/count).
#define WAVES 4   // waves per block (128 threads); each wave owns 2 M-tiles
__global__ __launch_bounds__(128) void mlp_ln_kernel(
    const _Float16* __restrict__ X, int Ksteps,            // K = Ksteps*32
    const _Float16* __restrict__ W1p, const float* __restrict__ b1,
    const _Float16* __restrict__ W2p, const float* __restrict__ b2,
    const float* __restrict__ g, const float* __restrict__ beta,
    float* __restrict__ out, const float* __restrict__ resid,
    float* aggS, float* aggMx, float* aggMn, float* aggC,
    const int* __restrict__ recv, int ntiles) {

    __shared__ __align__(16) _Float16 hbuf[WAVES][2][16][136];  // padded vs bank conflicts

    const int wave = threadIdx.x >> 5;
    const int lane = threadIdx.x & 31;
    const int t0   = (blockIdx.x * WAVES + wave) * 2;
    if (t0 >= ntiles) return;                  // wave-independent; no block barriers used
    const bool valid1 = (t0 + 1) < ntiles;
    const int hs  = (lane < 16) ? 0 : 1;       // half-wave select
    const int col = lane & 15;
    const int K   = Ksteps * 32;

    int rowBase[2];
    rowBase[0] = t0 * 16;
    rowBase[1] = (valid1 ? t0 + 1 : t0) * 16;  // clamp: recompute tile0, skip epilogue

    // ---------------- layer 1: [32 x K] @ [K x 128] -> acc[2][8] (16x16 tiles)
    v8f acc[2][8];
#pragma unroll
    for (int u = 0; u < 2; ++u)
#pragma unroll
        for (int t = 0; t < 8; ++t)
#pragma unroll
            for (int v = 0; v < 8; ++v) acc[u][t][v] = 0.0f;

    const _Float16* xr0 = X + ((size_t)rowBase[0] + col) * K + hs * 8;
    const _Float16* xr1 = X + ((size_t)rowBase[1] + col) * K + hs * 8;

    for (int ks = 0; ks < Ksteps; ++ks) {
        if (ks + 2 < Ksteps) __builtin_prefetch(xr0 + (ks + 2) * 32, 0, 3);
        union { v16h v; v8h h[2]; } a0, a1;
        a0.h[0] = *(const v8h*)(xr0 + ks * 32);        // K = ks*32 + hs*8 + 0..7
        a0.h[1] = *(const v8h*)(xr0 + ks * 32 + 16);   // K = ks*32 + hs*8 + 16..23
        a1.h[0] = *(const v8h*)(xr1 + ks * 32);
        a1.h[1] = *(const v8h*)(xr1 + ks * 32 + 16);
        const v16h* wf = (const v16h*)W1p + ((size_t)ks * 8) * 32 + lane;
        v16h bf[8];
#pragma unroll
        for (int t = 0; t < 8; ++t) bf[t] = wf[t * 32];   // load clause; progressive waits
#pragma unroll
        for (int t = 0; t < 8; ++t) {
            acc[0][t] = __builtin_amdgcn_wmma_f32_16x16x32_f16(
                false, a0.v, false, bf[t], (short)0, acc[0][t], false, false);
            acc[1][t] = __builtin_amdgcn_wmma_f32_16x16x32_f16(
                false, a1.v, false, bf[t], (short)0, acc[1][t], false, false);
        }
    }

    // bias + relu, transpose C-layout -> A-layout through wave-private LDS
#pragma unroll
    for (int u = 0; u < 2; ++u)
#pragma unroll
        for (int t = 0; t < 8; ++t) {
            float bb = b1[t * 16 + col];
#pragma unroll
            for (int v = 0; v < 8; ++v) {
                float h = acc[u][t][v] + bb;
                h = (h > 0.0f) ? h : 0.0f;
                hbuf[wave][u][v + hs * 8][t * 16 + col] = (_Float16)h;
            }
        }
    __asm__ volatile("s_wait_dscnt 0x0" ::: "memory");   // wave-local LDS visibility

    // ---------------- layer 2: [32 x 128] @ [128 x 128]
    v8f acc2[2][8];
#pragma unroll
    for (int u = 0; u < 2; ++u)
#pragma unroll
        for (int t = 0; t < 8; ++t)
#pragma unroll
            for (int v = 0; v < 8; ++v) acc2[u][t][v] = 0.0f;

#pragma unroll
    for (int ks = 0; ks < 4; ++ks) {
        const v16h* wf = (const v16h*)W2p + ((size_t)ks * 8) * 32 + lane;
        v16h bf[8];
#pragma unroll
        for (int t = 0; t < 8; ++t) bf[t] = wf[t * 32];
        union { v16h v; v8h h[2]; } a0, a1;
        a0.h[0] = *(const v8h*)(&hbuf[wave][0][col][ks * 32 + hs * 8]);
        a0.h[1] = *(const v8h*)(&hbuf[wave][0][col][ks * 32 + hs * 8 + 16]);
        a1.h[0] = *(const v8h*)(&hbuf[wave][1][col][ks * 32 + hs * 8]);
        a1.h[1] = *(const v8h*)(&hbuf[wave][1][col][ks * 32 + hs * 8 + 16]);
#pragma unroll
        for (int t = 0; t < 8; ++t) {
            acc2[0][t] = __builtin_amdgcn_wmma_f32_16x16x32_f16(
                false, a0.v, false, bf[t], (short)0, acc2[0][t], false, false);
            acc2[1][t] = __builtin_amdgcn_wmma_f32_16x16x32_f16(
                false, a1.v, false, bf[t], (short)0, acc2[1][t], false, false);
        }
    }

    // ---------------- epilogue per tile: bias2, LN, PNA atomics, residual, store
    const bool doAgg = (aggS != nullptr);
#pragma unroll
    for (int u = 0; u < 2; ++u) {
        if (u == 1 && !valid1) break;

#pragma unroll
        for (int t = 0; t < 8; ++t) {
            float bb = b2[t * 16 + col];
#pragma unroll
            for (int v = 0; v < 8; ++v) acc2[u][t][v] += bb;
        }

        // LayerNorm over feature dim (row m = v + hs*8 spread over 16 lanes x 8 tiles)
        float rs[8], rss[8];
#pragma unroll
        for (int v = 0; v < 8; ++v) {
            float s = 0.0f, ss = 0.0f;
#pragma unroll
            for (int t = 0; t < 8; ++t) { float y = acc2[u][t][v]; s += y; ss += y * y; }
            rs[v] = s; rss[v] = ss;
        }
        for (int m = 1; m < 16; m <<= 1) {
#pragma unroll
            for (int v = 0; v < 8; ++v) {
                rs[v]  += __shfl_xor(rs[v],  m, 32);
                rss[v] += __shfl_xor(rss[v], m, 32);
            }
        }
        float mean[8], rstd[8];
#pragma unroll
        for (int v = 0; v < 8; ++v) {
            mean[v] = rs[v] * (1.0f / 128.0f);
            float var = rss[v] * (1.0f / 128.0f) - mean[v] * mean[v];
            rstd[v] = rsqrtf(var + 1e-5f);
        }

#pragma unroll
        for (int t = 0; t < 8; ++t) {
            int n = t * 16 + col;
            float gn = g[n], bn = beta[n];
#pragma unroll
            for (int v = 0; v < 8; ++v) {
                int rowg = rowBase[u] + v + hs * 8;
                float val = (acc2[u][t][v] - mean[v]) * rstd[v] * gn + bn;  // MLP output
                if (doAgg) {
                    int r = recv[rowg];
                    size_t ai = (size_t)r * D_ + n;
                    atomicAdd(&aggS[ai], val);
                    atomicMaxF(&aggMx[ai], val);
                    atomicMinF(&aggMn[ai], val);
                }
                float o = val;
                if (resid) o += resid[(size_t)rowg * D_ + n];
                out[(size_t)rowg * D_ + n] = o;
            }
        }
        if (doAgg && col == 0) {   // one count per edge row (lanes 0 and 16 cover 16 rows)
#pragma unroll
            for (int v = 0; v < 8; ++v) {
                int rowg = rowBase[u] + v + hs * 8;
                atomicAdd(&aggC[recv[rowg]], 1.0f);
            }
        }
    }
}

// ---------------------------------------------------------------- launcher
extern "C" void kernel_launch(void* const* d_in, const int* in_sizes, int n_in,
                              void* d_out, int out_size, void* d_ws, size_t ws_size,
                              hipStream_t stream) {
    const float* mesh  = (const float*)d_in[0];
    const float* hyper = (const float*)d_in[1];
    const float* efeat[4] = {(const float*)d_in[2], (const float*)d_in[3],
                             (const float*)d_in[4], (const float*)d_in[5]};
    const int* sidx[4] = {(const int*)d_in[6], (const int*)d_in[8],
                          (const int*)d_in[10], (const int*)d_in[12]};
    const int* ridx[4] = {(const int*)d_in[7], (const int*)d_in[9],
                          (const int*)d_in[11], (const int*)d_in[13]};
    const int Ecnt[4] = {EM_, EU_, EI_, ED_};
    auto P = [&](int m, int t) { return (const float*)d_in[14 + m * 6 + t]; };
    // params order: edge_mesh, edge_up, edge_inter, edge_down,
    //               node_cross(4), hyper_up(5), hyper_cross(6), node_down(7)
    // per-mlp tensors: W1(0), b1(1), W2(2), b2(3), g(4), beta(5)

    // ---- workspace carve-out
    char* base = (char*)d_ws;
    size_t off = 0;
    auto alloc = [&](size_t bytes) -> void* {
        off = (off + 255) & ~(size_t)255;
        void* p = base + off;
        off += bytes;
        return p;
    };

    const int Kmlp[8] = {384, 384, 384, 384, 2176, 640, 640, 640};
    _Float16* w1p[8]; _Float16* w2p[8];
    for (int m = 0; m < 8; ++m) {
        w1p[m] = (_Float16*)alloc((size_t)Kmlp[m] * D_ * 2);
        w2p[m] = (_Float16*)alloc((size_t)H_ * D_ * 2);
    }
    float *AS[4], *AMX[4], *AMN[4], *AC[4];
    for (int j = 0; j < 4; ++j) {
        AS[j]  = (float*)alloc((size_t)NTOT_ * D_ * 4);
        AMX[j] = (float*)alloc((size_t)NTOT_ * D_ * 4);
        AMN[j] = (float*)alloc((size_t)NTOT_ * D_ * 4);
        AC[j]  = (float*)alloc((size_t)NTOT_ * 4);
    }
    size_t bigBytes = (size_t)NM_ * 2176 * 2;           // >= EM_*384*2
    if ((size_t)EM_ * 384 * 2 > bigBytes) bigBytes = (size_t)EM_ * 384 * 2;
    _Float16* bigX = (_Float16*)alloc(bigBytes);        // edge staging, later Xc
    _Float16* Xhu = (_Float16*)alloc((size_t)NH_ * 640 * 2);
    _Float16* Xhc = (_Float16*)alloc((size_t)NH_ * 640 * 2);
    _Float16* Xnd = (_Float16*)alloc((size_t)NM_ * 640 * 2);
    float* updCross = (float*)alloc((size_t)NM_ * D_ * 4);
    float* hypUp    = (float*)alloc((size_t)NH_ * D_ * 4);

    // ---- 1. pack weights to f16 B-fragment layout
    for (int m = 0; m < 8; ++m) {
        int t1 = Kmlp[m] * D_;
        pack_w<<<(t1 + 255) / 256, 256, 0, stream>>>(P(m, 0), Kmlp[m], w1p[m]);
        pack_w<<<(H_ * D_ + 255) / 256, 256, 0, stream>>>(P(m, 2), H_, w2p[m]);
    }

    // ---- 2. init PNA buffers (every launch; deterministic)
    for (int j = 0; j < 4; ++j) {
        int cnt = NTOT_ * D_;
        fill_u32<<<(cnt + 255) / 256, 256, 0, stream>>>((unsigned*)AS[j],  cnt, 0u);
        fill_u32<<<(cnt + 255) / 256, 256, 0, stream>>>((unsigned*)AMX[j], cnt, 0xFF800000u); // -inf
        fill_u32<<<(cnt + 255) / 256, 256, 0, stream>>>((unsigned*)AMN[j], cnt, 0x7F800000u); // +inf
        fill_u32<<<(NTOT_ + 255) / 256, 256, 0, stream>>>((unsigned*)AC[j], NTOT_, 0u);
    }

    // ---- output layout
    float* outMesh  = (float*)d_out;
    float* outHyper = outMesh + (size_t)NM_ * D_;
    float* outEdge[4];
    {
        size_t eo = (size_t)NTOT_ * D_;
        for (int j = 0; j < 4; ++j) { outEdge[j] = outMesh + eo; eo += (size_t)Ecnt[j] * D_; }
    }

    AggPtrs agg[4];
    for (int j = 0; j < 4; ++j) agg[j] = AggPtrs{AS[j], AMX[j], AMN[j], AC[j]};

    const int TPB = WAVES * 2;   // tiles per block (4 waves x 2 tiles)

    // ---- 3. edge MLPs (fused LN + residual-to-output + PNA atomics)
    for (int j = 0; j < 4; ++j) {
        int tot = Ecnt[j] * 384;
        build_edge_x<<<(tot + 255) / 256, 256, 0, stream>>>(
            mesh, hyper, efeat[j], sidx[j], ridx[j], Ecnt[j], bigX);
        int nt = Ecnt[j] / 16;
        mlp_ln_kernel<<<(nt + TPB - 1) / TPB, 128, 0, stream>>>(
            bigX, 384 / 32, w1p[j], P(j, 1), w2p[j], P(j, 3), P(j, 4), P(j, 5),
            outEdge[j], efeat[j], AS[j], AMX[j], AMN[j], AC[j], ridx[j], nt);
    }

    // ---- 4. node_cross (mesh): K = 128 + 4*512 = 2176
    {
        int tot = NM_ * 2176;
        build_xc<<<(tot + 255) / 256, 256, 0, stream>>>(
            mesh, agg[0], agg[1], agg[2], agg[3], bigX);
        int nt = NM_ / 16;
        mlp_ln_kernel<<<(nt + TPB - 1) / TPB, 128, 0, stream>>>(
            bigX, 2176 / 32, w1p[4], P(4, 1), w2p[4], P(4, 3), P(4, 4), P(4, 5),
            updCross, nullptr, nullptr, nullptr, nullptr, nullptr, nullptr, nt);
    }

    // ---- 5. hyper_up: [hyper | a_up[NM:]] -> hypUp
    {
        int tot = NH_ * 640;
        build_x_single<<<(tot + 255) / 256, 256, 0, stream>>>(hyper, agg[1], NM_, NH_, Xhu);
        int nt = NH_ / 16;
        mlp_ln_kernel<<<(nt + TPB - 1) / TPB, 128, 0, stream>>>(
            Xhu, 640 / 32, w1p[5], P(5, 1), w2p[5], P(5, 3), P(5, 4), P(5, 5),
            hypUp, nullptr, nullptr, nullptr, nullptr, nullptr, nullptr, nt);
    }

    // ---- 6. hyper_cross: [hypUp | a_inter[NM:]] -> out_hyper (+hyper residual)
    {
        int tot = NH_ * 640;
        build_x_single<<<(tot + 255) / 256, 256, 0, stream>>>(hypUp, agg[2], NM_, NH_, Xhc);
        int nt = NH_ / 16;
        mlp_ln_kernel<<<(nt + TPB - 1) / TPB, 128, 0, stream>>>(
            Xhc, 640 / 32, w1p[6], P(6, 1), w2p[6], P(6, 3), P(6, 4), P(6, 5),
            outHyper, hyper, nullptr, nullptr, nullptr, nullptr, nullptr, nt);
    }

    // ---- 7. node_down: [updCross | a_down[:NM]] -> out_mesh (+mesh residual)
    {
        int tot = NM_ * 640;
        build_x_single<<<(tot + 255) / 256, 256, 0, stream>>>(updCross, agg[3], 0, NM_, Xnd);
        int nt = NM_ / 16;
        mlp_ln_kernel<<<(nt + TPB - 1) / TPB, 128, 0, stream>>>(
            Xnd, 640 / 32, w1p[7], P(7, 1), w2p[7], P(7, 3), P(7, 4), P(7, 5),
            outMesh, mesh, nullptr, nullptr, nullptr, nullptr, nullptr, nt);
    }
}
